// BiDense_6047313952873
// MI455X (gfx1250) — 8-bit matrix operand order repaired and checked by reading, compiled, not run on hardware
//
#include <hip/hip_runtime.h>
#include <hip/hip_bf16.h>
#include <cstdint>


typedef __attribute__((ext_vector_type(8))) int v8i;

#define D_DIM 4096
#define F_DIM 16384
#define M_DIM 8192
#define FLT_EPS 1.1920928955078125e-07f

__global__ __launch_bounds__(256) void binx_kernel(const float* __restrict__ x,
                                                   int8_t* __restrict__ xb,
                                                   float* __restrict__ bound_x) {
  const int m = blockIdx.x;
  const int t = threadIdx.x;
  const float* row = x + (size_t)m * D_DIM;
  int8_t* orow = xb + (size_t)m * D_DIM;
  float mx = 0.f;
#pragma unroll
  for (int i = 0; i < 4; ++i) {
    const int v4 = t + i * 256;
    float4 v = ((const float4*)row)[v4];
    char4 c;
    c.x = (v.x >= 0.f) ? 1 : -1;
    c.y = (v.y >= 0.f) ? 1 : -1;
    c.z = (v.z >= 0.f) ? 1 : -1;
    c.w = (v.w >= 0.f) ? 1 : -1;
    mx = fmaxf(mx, fmaxf(fmaxf(fabsf(v.x), fabsf(v.y)),
                         fmaxf(fabsf(v.z), fabsf(v.w))));
    ((char4*)orow)[v4] = c;
  }
  __shared__ float red[256];
  red[t] = mx;
  __syncthreads();
  for (int s = 128; s > 0; s >>= 1) {
    if (t < s) red[t] = fmaxf(red[t], red[t + s]);
    __syncthreads();
  }
  if (t == 0) bound_x[m] = red[0] + FLT_EPS;
}

__global__ __launch_bounds__(256) void bk_init_kernel(unsigned* __restrict__ bk) {
  bk[blockIdx.x * 256 + threadIdx.x] = 0u;
}

__global__ __launch_bounds__(256) void binw_kernel(const float* __restrict__ w,
                                                   int8_t* __restrict__ wT,
                                                   unsigned* __restrict__ bk_bits) {
  __shared__ __align__(16) int8_t tile[64][80];
  __shared__ unsigned cmax[64];
  const int t = threadIdx.x;
  if (t < 64) cmax[t] = 0u;
  __syncthreads();
  const int f0 = blockIdx.x * 64;
  const int d0 = blockIdx.y * 64;
  const int fl = t & 63;
  const int dl0 = t >> 6;
  float lm = 0.f;
#pragma unroll
  for (int r = 0; r < 16; ++r) {
    const int dl = dl0 + r * 4;
    float v = w[(size_t)(d0 + dl) * F_DIM + f0 + fl];
    lm = fmaxf(lm, fabsf(v));
    tile[fl][dl] = (v >= 0.f) ? 1 : -1;
  }
  atomicMax(&cmax[fl], __float_as_uint(lm));
  __syncthreads();
  const int fo = t >> 2;
  const int doff = (t & 3) * 16;
  int4 val = *(const int4*)&tile[fo][doff];
  *(int4*)(wT + (size_t)(f0 + fo) * D_DIM + d0 + doff) = val;
  if (t < 64) atomicMax(&bk_bits[f0 + t], cmax[t]);
}

__global__ __launch_bounds__(256) void bk_fin_kernel(unsigned* __restrict__ bits,
                                                     float* __restrict__ bound_k) {
  const int f = blockIdx.x * 256 + threadIdx.x;
  float b = __uint_as_float(bits[f]) + FLT_EPS;
  bound_k[f] = b;
}

__global__ __launch_bounds__(256) void gemm_kernel(const int8_t* __restrict__ A,
                                                   const int8_t* __restrict__ Bt,
                                                   const float* __restrict__ bound_x,
                                                   const float* __restrict__ bound_k,
                                                   const float* __restrict__ bias,
                                                   float* __restrict__ out) {
  __shared__ __align__(16) int8_t As[2][128][80];
  __shared__ __align__(16) int8_t Bs[2][128][80];

  const int t = threadIdx.x;
  const int lane = t & 31;
  const int wave = t >> 5;
  const int wm = (wave >> 2) * 64;
  const int wn = (wave & 3) * 32;
  const int m0 = blockIdx.y * 128;
  const int n0 = blockIdx.x * 128;

  const int lr = t >> 1;
  const int lc = (t & 1) * 32;

  const int8_t* aptr = A + (size_t)(m0 + lr) * D_DIM + lc;
  const int8_t* bptr = Bt + (size_t)(n0 + lr) * D_DIM + lc;

  unsigned ldsA[2], ldsB[2];
  ldsA[0] = (unsigned)(uintptr_t)&As[0][lr][lc];
  ldsA[1] = (unsigned)(uintptr_t)&As[1][lr][lc];
  ldsB[0] = (unsigned)(uintptr_t)&Bs[0][lr][lc];
  ldsB[1] = (unsigned)(uintptr_t)&Bs[1][lr][lc];

  auto issue_stage = [&](int buf, int kt) {
    const unsigned long long ga = (unsigned long long)(uintptr_t)(aptr + (size_t)kt * 64);
    const unsigned long long gb = (unsigned long long)(uintptr_t)(bptr + (size_t)kt * 64);
    asm volatile("global_load_async_to_lds_b128 %0, %1, off"
                 :: "v"(ldsA[buf]), "v"(ga) : "memory");
    asm volatile("global_load_async_to_lds_b128 %0, %1, off offset:16"
                 :: "v"(ldsA[buf]), "v"(ga) : "memory");
    asm volatile("global_load_async_to_lds_b128 %0, %1, off"
                 :: "v"(ldsB[buf]), "v"(gb) : "memory");
    asm volatile("global_load_async_to_lds_b128 %0, %1, off offset:16"
                 :: "v"(ldsB[buf]), "v"(gb) : "memory");
  };

  v8i acc[4][2];
#pragma unroll
  for (int i = 0; i < 4; ++i)
#pragma unroll
    for (int j = 0; j < 2; ++j)
      acc[i][j] = (v8i){0, 0, 0, 0, 0, 0, 0, 0};

  const int a_row = (lane & 15);
  const int a_kb = (lane >> 4) * 16;
  const int b_n = (lane & 15);
  const int b_kh = (lane >> 4) * 16;

  union AF { unsigned long long q[4]; v8i v; };
  union BF { int4 x[2]; v8i v; };

  constexpr int NT = D_DIM / 64;
  issue_stage(0, 0);

#pragma unroll 1
  for (int kt = 0; kt < NT; ++kt) {
    asm volatile("s_wait_asynccnt 0x0" ::: "memory");
    __syncthreads();
    if (kt + 1 < NT) issue_stage((kt + 1) & 1, kt + 1);

    const int buf = kt & 1;
    AF af[4];
#pragma unroll
    for (int i = 0; i < 4; ++i) {
      const int8_t* p = &As[buf][wm + i * 16 + a_row][a_kb];
      af[i].q[0] = *(const unsigned long long*)(p);
      af[i].q[1] = *(const unsigned long long*)(p + 8);
      af[i].q[2] = *(const unsigned long long*)(p + 32);
      af[i].q[3] = *(const unsigned long long*)(p + 40);
    }
    BF bf[2];
#pragma unroll
    for (int j = 0; j < 2; ++j) {
      const int8_t* p = &Bs[buf][wn + j * 16 + b_n][b_kh];
      bf[j].x[0] = *(const int4*)(p);
      bf[j].x[1] = *(const int4*)(p + 32);
    }
#pragma unroll
    for (int i = 0; i < 4; ++i)
#pragma unroll
      for (int j = 0; j < 2; ++j)
        acc[i][j] = __builtin_amdgcn_wmma_i32_16x16x64_iu8(
             true, af[i].v,  true, bf[j].v, acc[i][j],
             false,  false);
  }

  const int mlane = (lane >> 4) * 8;
#pragma unroll
  for (int i = 0; i < 4; ++i) {
#pragma unroll
    for (int j = 0; j < 2; ++j) {
      const int nn = n0 + wn + j * 16 + (lane & 15);
      const float bk = bound_k[nn];
      const float bs = bias[nn];
#pragma unroll
      for (int r = 0; r < 8; ++r) {
        const int mm = m0 + wm + i * 16 + mlane + r;
        const float s = 0.25f * bound_x[mm] * bk;
        out[(size_t)mm * F_DIM + nn] = s * (float)acc[i][j][r] + bs;
      }
    }
  }
}

extern "C" void kernel_launch(void* const* d_in, const int* in_sizes, int n_in,
                              void* d_out, int out_size, void* d_ws, size_t ws_size,
                              hipStream_t stream) {
  const float* x = (const float*)d_in[0];
  const float* w = (const float*)d_in[1];
  const float* bias = (const float*)d_in[2];
  float* out = (float*)d_out;

  int8_t* xb = (int8_t*)d_ws;
  int8_t* wT = xb + (size_t)M_DIM * D_DIM;
  float* bx = (float*)(wT + (size_t)F_DIM * D_DIM);
  unsigned* bk = (unsigned*)(bx + M_DIM);

  binx_kernel<<<M_DIM, 256, 0, stream>>>(x, xb, bx);
  bk_init_kernel<<<F_DIM / 256, 256, 0, stream>>>(bk);
  binw_kernel<<<dim3(F_DIM / 64, D_DIM / 64), 256, 0, stream>>>(w, wT, bk);
  bk_fin_kernel<<<F_DIM / 256, 256, 0, stream>>>(bk, (float*)bk);
  gemm_kernel<<<dim3(F_DIM / 128, M_DIM / 128), 256, 0, stream>>>(
      xb, wT, bx, (const float*)bk, bias, out);
}
